// MultiHeadAttentionSegment_52321291600194
// MI455X (gfx1250) — matrix operand order repaired and checked by reading, not compiled, not run on hardware
//
#include <hip/hip_runtime.h>

// ---------------------------------------------------------------------------
// Problem constants (B=2, S=2048, D=1024, H=16, DH=64)
// ---------------------------------------------------------------------------
#define BB 2
#define SS 2048
#define DD 1024
#define HH 16
#define DHH 64

typedef __bf16 bf16;
typedef __attribute__((ext_vector_type(16))) __bf16 v16bf;
typedef __attribute__((ext_vector_type(8)))  __bf16 v8bf;
typedef __attribute__((ext_vector_type(8)))  float  v8f;

// ---------------------------------------------------------------------------
// WMMA helper — CDNA5 v_wmma_f32_16x16x32_bf16 (wave32)
// ---------------------------------------------------------------------------
__device__ __forceinline__ v8f wmma_bf16(v16bf a, v16bf b, v8f c) {
    // (neg_a, A, neg_b, B, c_mod, C, reuse_a, reuse_b)
    return __builtin_amdgcn_wmma_f32_16x16x32_bf16(false, a, false, b,
                                                   (short)0, c, false, false);
}

// A fragment, 16x32 bf16 (M x K). ISA layout: lane m=lane&15;
// VGPRs0-3: K = half*8 + 0..7 ; VGPRs4-7: K = 16 + half*8 + 0..7
__device__ __forceinline__ v16bf load_frag_a(const bf16* base, int ld, int lane) {
    const int m = lane & 15, hlf = lane >> 4;
    const bf16* p = base + m * ld + hlf * 8;
    v8bf lo = *(const v8bf*)(p);
    v8bf hi = *(const v8bf*)(p + 16);
    return __builtin_shufflevector(lo, hi, 0,1,2,3,4,5,6,7,8,9,10,11,12,13,14,15);
}

// B fragment, 32x16 bf16 (K x N), stored as [n][k] (col-of-B contiguous in K).
// ISA layout: lane n=lane&15; VGPRs0-3: K = half*8 + 0..7 ; VGPRs4-7: K = 16 + half*8 + 0..7
__device__ __forceinline__ v16bf load_frag_b(const bf16* base, int ld, int lane) {
    const int n = lane & 15, hlf = lane >> 4;
    const bf16* p = base + n * ld + hlf * 8;
    v8bf lo = *(const v8bf*)(p);
    v8bf hi = *(const v8bf*)(p + 16);
    return __builtin_shufflevector(lo, hi, 0,1,2,3,4,5,6,7,8,9,10,11,12,13,14,15);
}

// CDNA5 async global->LDS copy, 16B per lane, GVS addressing:
//   mem = SGPR_base + VGPR_off32 ; LDS dest = per-lane VGPR offset.
// Tracked by ASYNCcnt (s_wait_asynccnt), data never passes through VGPRs.
__device__ __forceinline__ void async_copy16(unsigned lds_off,
                                             const void* gbase,
                                             unsigned byte_off) {
    asm volatile("global_load_async_to_lds_b128 %0, %1, %2"
                 :: "v"(lds_off), "v"(byte_off),
                    "s"((unsigned long long)(uintptr_t)gbase)
                 : "memory");
}
__device__ __forceinline__ void wait_async0() {
    asm volatile("s_wait_asynccnt 0x0" ::: "memory");
}

// ---------------------------------------------------------------------------
// Kernel 1: fused QKV projection.  C[m,n] = X[m,:] . W[n,:] + bias[n]
// Output written bf16 into head-split layout [B][H][S][DH].
// 128x128x32 block tiles, 8 waves in 2x4, each wave 64x32 (4x2 WMMA accs).
// Register-prefetch double buffering: next f32 slab loads during WMMAs.
// ---------------------------------------------------------------------------
#define BM 128
#define BN 128
#define BK 32
#define LDG (BK + 8)   // padded LDS row (bf16 elems), keeps 16B alignment

__global__ void __launch_bounds__(256)
qkv_proj_kernel(const float* __restrict__ xq, const float* __restrict__ xk,
                const float* __restrict__ xv,
                const float* __restrict__ wq, const float* __restrict__ wk,
                const float* __restrict__ wv,
                const float* __restrict__ bq, const float* __restrict__ bk,
                const float* __restrict__ bv,
                bf16* __restrict__ dq, bf16* __restrict__ dk,
                bf16* __restrict__ dv)
{
    const float* X; const float* W; const float* bias; bf16* Dst;
    if (blockIdx.z == 0)      { X = xq; W = wq; bias = bq; Dst = dq; }
    else if (blockIdx.z == 1) { X = xk; W = wk; bias = bk; Dst = dk; }
    else                      { X = xv; W = wv; bias = bv; Dst = dv; }

    __shared__ __align__(16) bf16 As[BM][LDG];
    __shared__ __align__(16) bf16 Bs[BN][LDG];

    const int tid  = threadIdx.x;
    const int lane = tid & 31;
    const int wv_  = tid >> 5;      // wave 0..7
    const int wm   = wv_ >> 2;      // 0..1
    const int wn   = wv_ & 3;       // 0..3
    const int m0   = blockIdx.y * BM;
    const int n0   = blockIdx.x * BN;

    v8f acc[4][2];
    #pragma unroll
    for (int mi = 0; mi < 4; ++mi)
        #pragma unroll
        for (int ni = 0; ni < 2; ++ni) acc[mi][ni] = {};

    // prefetch slab k0=0 into registers
    float4 ra[4], rb[4];
    #pragma unroll
    for (int i = 0; i < 4; ++i) {
        int id  = tid + i * 256;
        int row = id >> 3;
        int kq  = (id & 7) * 4;
        ra[i] = *(const float4*)(X + (size_t)(m0 + row) * DD + kq);
        rb[i] = *(const float4*)(W + (size_t)(n0 + row) * DD + kq);
    }

    for (int k0 = 0; k0 < DD; k0 += BK) {
        // registers -> LDS (f32 -> bf16)
        #pragma unroll
        for (int i = 0; i < 4; ++i) {
            int id  = tid + i * 256;
            int row = id >> 3;
            int kq  = (id & 7) * 4;
            bf16* da = &As[row][kq];
            da[0] = (bf16)ra[i].x; da[1] = (bf16)ra[i].y;
            da[2] = (bf16)ra[i].z; da[3] = (bf16)ra[i].w;
            bf16* db = &Bs[row][kq];
            db[0] = (bf16)rb[i].x; db[1] = (bf16)rb[i].y;
            db[2] = (bf16)rb[i].z; db[3] = (bf16)rb[i].w;
        }
        __syncthreads();

        // kick off next slab's global loads; they retire during the WMMAs
        if (k0 + BK < DD) {
            #pragma unroll
            for (int i = 0; i < 4; ++i) {
                int id  = tid + i * 256;
                int row = id >> 3;
                int kq  = (id & 7) * 4;
                ra[i] = *(const float4*)(X + (size_t)(m0 + row) * DD + k0 + BK + kq);
                rb[i] = *(const float4*)(W + (size_t)(n0 + row) * DD + k0 + BK + kq);
            }
        }

        v16bf af[4], bfv[2];
        #pragma unroll
        for (int mi = 0; mi < 4; ++mi)
            af[mi] = load_frag_a(&As[wm * 64 + mi * 16][0], LDG, lane);
        #pragma unroll
        for (int ni = 0; ni < 2; ++ni)
            bfv[ni] = load_frag_b(&Bs[wn * 32 + ni * 16][0], LDG, lane);
        #pragma unroll
        for (int mi = 0; mi < 4; ++mi)
            #pragma unroll
            for (int ni = 0; ni < 2; ++ni)
                acc[mi][ni] = wmma_bf16(af[mi], bfv[ni], acc[mi][ni]);
        __syncthreads();
    }

    // epilogue: bias add, convert to bf16, scatter into [B][H][S][DH]
    const int nl = lane & 15, hlf = lane >> 4;
    #pragma unroll
    for (int ni = 0; ni < 2; ++ni) {
        int n = n0 + wn * 32 + ni * 16 + nl;
        float bia = bias[n];
        int h = n >> 6, dh = n & 63;
        #pragma unroll
        for (int mi = 0; mi < 4; ++mi) {
            #pragma unroll
            for (int r = 0; r < 8; ++r) {
                int m  = m0 + wm * 64 + mi * 16 + r + hlf * 8;
                int bb = m >> 11;            // / S
                int s  = m & (SS - 1);
                size_t di = (((size_t)bb * HH + h) * SS + s) * DHH + dh;
                Dst[di] = (bf16)(acc[mi][ni][r] + bia);
            }
        }
    }
}

// ---------------------------------------------------------------------------
// Kernel 2: flash attention. One block = one (b,h) x 128 q-rows.
// Each wave owns 16 q rows; K-blocks of 64 keys; online softmax.
// K/V tiles double-buffered and filled with async global->LDS copies:
//   iter kb:  wait(kb data) ; barrier ; issue(kb+1) ; Vraw->Vt transpose ;
//             barrier ; WMMA compute(kb)
// Buffer safety: issue(kb+1) targets buf (kb+1)&1 whose last readers were
// compute(kb-1)/transpose(kb-1) — both strictly before this iter's top
// barrier. Vt needs no double buffer: the top barrier of kb+1 separates
// compute(kb) from transpose(kb+1).
// ---------------------------------------------------------------------------
#define LDT 72             // padded LDS row for 64-wide tiles
#define NKB (SS / 64)      // 32 key blocks

__global__ void __launch_bounds__(256)
attn_kernel(const bf16* __restrict__ Q, const bf16* __restrict__ K,
            const bf16* __restrict__ V, bf16* __restrict__ O)
{
    __shared__ __align__(16) bf16 Ks[2][64][LDT];
    __shared__ __align__(16) bf16 Vraw[2][64][LDT];    // V rows as in memory
    __shared__ __align__(16) bf16 Vt[64][LDT];         // V transposed: [dh][k]
    __shared__ __align__(16) bf16 Ps[8][16][LDT];      // per-wave P tile [q][k]

    const int tid  = threadIdx.x;
    const int lane = tid & 31;
    const int wv_  = tid >> 5;
    const int nl   = lane & 15, hlf = lane >> 4;
    const int bh   = blockIdx.y;                       // b*H + h
    const int q0   = blockIdx.x * 128 + wv_ * 16;
    const size_t base = (size_t)bh * SS * DHH;
    const bf16* Qb = Q + base;
    const bf16* Kb = K + base;
    const bf16* Vb = V + base;

    // this thread's two 16B staging chunks (row 0..63, col 0..56 step 8)
    const int id0 = tid,        row0 = id0 >> 3, c80 = (id0 & 7) * 8;
    const int id1 = tid + 256,  row1 = id1 >> 3, c81 = (id1 & 7) * 8;

    // prologue: async-stage block 0 into buffer 0
    {
        unsigned g0 = (unsigned)((row0 * DHH + c80) * sizeof(bf16));
        unsigned g1 = (unsigned)((row1 * DHH + c81) * sizeof(bf16));
        async_copy16((unsigned)(uintptr_t)&Ks[0][row0][c80],   Kb, g0);
        async_copy16((unsigned)(uintptr_t)&Vraw[0][row0][c80], Vb, g0);
        async_copy16((unsigned)(uintptr_t)&Ks[0][row1][c81],   Kb, g1);
        async_copy16((unsigned)(uintptr_t)&Vraw[0][row1][c81], Vb, g1);
    }

    // Q fragments resident in VGPRs for the whole kernel (dh 0..31 / 32..63)
    v16bf aq0 = load_frag_a(Qb + (size_t)q0 * DHH,      DHH, lane);
    v16bf aq1 = load_frag_a(Qb + (size_t)q0 * DHH + 32, DHH, lane);

    v8f o[4];
    #pragma unroll
    for (int t = 0; t < 4; ++t) o[t] = {};
    float row_max[8], row_sum[8];
    #pragma unroll
    for (int r = 0; r < 8; ++r) { row_max[r] = -1e30f; row_sum[r] = 0.f; }
    const float scale = 0.125f;   // 1/sqrt(64); applied inside exp (monotone)

    for (int kb = 0; kb < NKB; ++kb) {
        const int buf = kb & 1;

        wait_async0();            // block kb's tiles landed in LDS (own lanes)
        __syncthreads();          // ...and everyone else's; compute(kb-1) done

        // async-stage block kb+1 into the other buffer
        if (kb + 1 < NKB) {
            const int nb = buf ^ 1;
            const unsigned koff = (unsigned)((kb + 1) * 64 * DHH * sizeof(bf16));
            unsigned g0 = koff + (unsigned)((row0 * DHH + c80) * sizeof(bf16));
            unsigned g1 = koff + (unsigned)((row1 * DHH + c81) * sizeof(bf16));
            async_copy16((unsigned)(uintptr_t)&Ks[nb][row0][c80],   Kb, g0);
            async_copy16((unsigned)(uintptr_t)&Vraw[nb][row0][c80], Vb, g0);
            async_copy16((unsigned)(uintptr_t)&Ks[nb][row1][c81],   Kb, g1);
            async_copy16((unsigned)(uintptr_t)&Vraw[nb][row1][c81], Vb, g1);
        }

        // transpose V rows -> Vt[dh][k]
        {
            v8bf v0 = *(const v8bf*)(&Vraw[buf][row0][c80]);
            #pragma unroll
            for (int j = 0; j < 8; ++j) Vt[c80 + j][row0] = v0[j];
            v8bf v1 = *(const v8bf*)(&Vraw[buf][row1][c81]);
            #pragma unroll
            for (int j = 0; j < 8; ++j) Vt[c81 + j][row1] = v1[j];
        }
        __syncthreads();

        // scores: 4 tiles of 16 keys, each = 2 chained WMMAs over dh
        v8f c[4];
        #pragma unroll
        for (int j = 0; j < 4; ++j) {
            v16bf bk0 = load_frag_b(&Ks[buf][j * 16][0],  LDT, lane);
            v16bf bk1 = load_frag_b(&Ks[buf][j * 16][32], LDT, lane);
            v8f z = {};
            z    = wmma_bf16(aq0, bk0, z);
            c[j] = wmma_bf16(aq1, bk1, z);
        }

        // ---- online softmax (C layout: lane = key col, VGPR r = q row) ----
        float vmax[8];
        #pragma unroll
        for (int r = 0; r < 8; ++r)
            vmax[r] = fmaxf(fmaxf(c[0][r], c[1][r]), fmaxf(c[2][r], c[3][r]));
        #pragma unroll
        for (int r = 0; r < 8; ++r) {
            #pragma unroll
            for (int mk = 1; mk < 16; mk <<= 1)
                vmax[r] = fmaxf(vmax[r], __shfl_xor(vmax[r], mk, 32));
        }
        float alpha[8];
        #pragma unroll
        for (int r = 0; r < 8; ++r) {
            float nm = fmaxf(row_max[r], vmax[r]);
            alpha[r] = __expf(scale * (row_max[r] - nm));
            row_max[r] = nm;
        }
        float ls[8];
        #pragma unroll
        for (int r = 0; r < 8; ++r) ls[r] = 0.f;
        #pragma unroll
        for (int j = 0; j < 4; ++j) {
            #pragma unroll
            for (int r = 0; r < 8; ++r) {
                float p = __expf(scale * (c[j][r] - row_max[r]));
                c[j][r] = p;
                ls[r]  += p;
            }
        }
        #pragma unroll
        for (int r = 0; r < 8; ++r) {
            #pragma unroll
            for (int mk = 1; mk < 16; mk <<= 1)
                ls[r] += __shfl_xor(ls[r], mk, 32);
            row_sum[r] = row_sum[r] * alpha[r] + ls[r];
        }
        #pragma unroll
        for (int t = 0; t < 4; ++t)
            #pragma unroll
            for (int r = 0; r < 8; ++r) o[t][r] *= alpha[r];

        // P tile -> per-wave LDS (A-operand layout for P@V)
        bf16* pw = &Ps[wv_][0][0];
        #pragma unroll
        for (int j = 0; j < 4; ++j)
            #pragma unroll
            for (int r = 0; r < 8; ++r)
                pw[(r + hlf * 8) * LDT + j * 16 + nl] = (bf16)c[j][r];

        // same-wave LDS RAW: DS ops are in-order per wave; fence the compiler
        asm volatile("s_wait_dscnt 0x0" ::: "memory");

        // O += P @ V   (A = P from LDS, B = Vt from LDS)
        #pragma unroll
        for (int jk = 0; jk < 2; ++jk) {
            v16bf ap = load_frag_a(pw + jk * 32, LDT, lane);
            #pragma unroll
            for (int t = 0; t < 4; ++t) {
                v16bf bv = load_frag_b(&Vt[t * 16][jk * 32], LDT, lane);
                o[t] = wmma_bf16(ap, bv, o[t]);
            }
        }
    }

    // normalize + write bf16 merged-head [B][S][D]
    const int b = bh >> 4, h = bh & 15;
    #pragma unroll
    for (int r = 0; r < 8; ++r) {
        float inv = 1.0f / row_sum[r];
        int s = q0 + r + hlf * 8;
        size_t rowbase = ((size_t)b * SS + s) * DD + h * DHH;
        #pragma unroll
        for (int t = 0; t < 4; ++t)
            O[rowbase + t * 16 + nl] = (bf16)(o[t][r] * inv);
    }
}

// ---------------------------------------------------------------------------
// Kernel 3: output projection. out[m,n] = Xo[m,:] . Wo[n,:] + bo[n]  (f32 out)
// Register-prefetch double buffering as in kernel 1.
// ---------------------------------------------------------------------------
__global__ void __launch_bounds__(256)
out_proj_kernel(const bf16* __restrict__ Xo, const float* __restrict__ Wo,
                const float* __restrict__ bo, float* __restrict__ out)
{
    __shared__ __align__(16) bf16 As[BM][LDG];
    __shared__ __align__(16) bf16 Bs[BN][LDG];

    const int tid  = threadIdx.x;
    const int lane = tid & 31;
    const int wv_  = tid >> 5;
    const int wm   = wv_ >> 2;
    const int wn   = wv_ & 3;
    const int m0   = blockIdx.y * BM;
    const int n0   = blockIdx.x * BN;

    v8f acc[4][2];
    #pragma unroll
    for (int mi = 0; mi < 4; ++mi)
        #pragma unroll
        for (int ni = 0; ni < 2; ++ni) acc[mi][ni] = {};

    uint4  ra[2];
    float4 rb[4];
    #pragma unroll
    for (int i = 0; i < 2; ++i) {
        int id = tid + i * 256, row = id >> 2, c8 = (id & 3) * 8;
        ra[i] = *(const uint4*)(Xo + (size_t)(m0 + row) * DD + c8);
    }
    #pragma unroll
    for (int i = 0; i < 4; ++i) {
        int id = tid + i * 256, row = id >> 3, kq = (id & 7) * 4;
        rb[i] = *(const float4*)(Wo + (size_t)(n0 + row) * DD + kq);
    }

    for (int k0 = 0; k0 < DD; k0 += BK) {
        #pragma unroll
        for (int i = 0; i < 2; ++i) {
            int id = tid + i * 256, row = id >> 2, c8 = (id & 3) * 8;
            *(uint4*)(&As[row][c8]) = ra[i];
        }
        #pragma unroll
        for (int i = 0; i < 4; ++i) {
            int id = tid + i * 256, row = id >> 3, kq = (id & 7) * 4;
            bf16* d = &Bs[row][kq];
            d[0] = (bf16)rb[i].x; d[1] = (bf16)rb[i].y;
            d[2] = (bf16)rb[i].z; d[3] = (bf16)rb[i].w;
        }
        __syncthreads();

        if (k0 + BK < DD) {
            #pragma unroll
            for (int i = 0; i < 2; ++i) {
                int id = tid + i * 256, row = id >> 2, c8 = (id & 3) * 8;
                ra[i] = *(const uint4*)(Xo + (size_t)(m0 + row) * DD + k0 + BK + c8);
            }
            #pragma unroll
            for (int i = 0; i < 4; ++i) {
                int id = tid + i * 256, row = id >> 3, kq = (id & 7) * 4;
                rb[i] = *(const float4*)(Wo + (size_t)(n0 + row) * DD + k0 + BK + kq);
            }
        }

        v16bf af[4], bfv[2];
        #pragma unroll
        for (int mi = 0; mi < 4; ++mi)
            af[mi] = load_frag_a(&As[wm * 64 + mi * 16][0], LDG, lane);
        #pragma unroll
        for (int ni = 0; ni < 2; ++ni)
            bfv[ni] = load_frag_b(&Bs[wn * 32 + ni * 16][0], LDG, lane);
        #pragma unroll
        for (int mi = 0; mi < 4; ++mi)
            #pragma unroll
            for (int ni = 0; ni < 2; ++ni)
                acc[mi][ni] = wmma_bf16(af[mi], bfv[ni], acc[mi][ni]);
        __syncthreads();
    }

    const int nl = lane & 15, hlf = lane >> 4;
    #pragma unroll
    for (int ni = 0; ni < 2; ++ni) {
        int n = n0 + wn * 32 + ni * 16 + nl;
        float bia = bo[n];
        #pragma unroll
        for (int mi = 0; mi < 4; ++mi) {
            #pragma unroll
            for (int r = 0; r < 8; ++r) {
                int m = m0 + wm * 64 + mi * 16 + r + hlf * 8;
                out[(size_t)m * DD + n] = acc[mi][ni][r] + bia;
            }
        }
    }
}

// ---------------------------------------------------------------------------
// Launch
// ---------------------------------------------------------------------------
extern "C" void kernel_launch(void* const* d_in, const int* in_sizes, int n_in,
                              void* d_out, int out_size, void* d_ws, size_t ws_size,
                              hipStream_t stream)
{
    const float* q   = (const float*)d_in[0];
    const float* k   = (const float*)d_in[1];
    const float* v   = (const float*)d_in[2];
    // d_in[3] = mask: reference's masked_fill result is discarded -> unused.
    const float* w_q = (const float*)d_in[4];
    const float* b_q = (const float*)d_in[5];
    const float* w_k = (const float*)d_in[6];
    const float* b_k = (const float*)d_in[7];
    const float* w_v = (const float*)d_in[8];
    const float* b_v = (const float*)d_in[9];
    const float* w_o = (const float*)d_in[10];
    const float* b_o = (const float*)d_in[11];

    const size_t nBHSD = (size_t)BB * HH * SS * DHH;   // 4,194,304 elems
    bf16* Qh = (bf16*)d_ws;
    bf16* Kh = Qh + nBHSD;
    bf16* Vh = Kh + nBHSD;
    bf16* Ob = Vh + nBHSD;                             // total 32 MB bf16

    dim3 blk(256);
    qkv_proj_kernel<<<dim3(DD / BN, (BB * SS) / BM, 3), blk, 0, stream>>>(
        q, k, v, w_q, w_k, w_v, b_q, b_k, b_v, Qh, Kh, Vh);
    attn_kernel<<<dim3(SS / 128, BB * HH), blk, 0, stream>>>(Qh, Kh, Vh, Ob);
    out_proj_kernel<<<dim3(DD / BN, (BB * SS) / BM), blk, 0, stream>>>(
        Ob, w_o, b_o, (float*)d_out);
}
